// GraphVAE_32504312496830
// MI455X (gfx1250) — compile-verified, matrix operand order checked by reading, not run on hardware
//
#include <hip/hip_runtime.h>
#include <hip/hip_bf16.h>

// ---------------------------------------------------------------------------
// GraphVAE forward for MI455X (gfx1250, wave32, WMMA).
// Big GEMMs use v_wmma_f32_16x16x32_f16 with LDS tiles stored in WMMA
// fragment order (2x ds_load_b128 per fragment). Irregular graph ops use
// VALU + global/LDS float atomics.
// ---------------------------------------------------------------------------

typedef __attribute__((ext_vector_type(16))) _Float16 v16h;
typedef __attribute__((ext_vector_type(8)))  _Float16 v8h;
typedef __attribute__((ext_vector_type(8)))  float    v8f;

#define BGR     2048            // number of graphs
#define NMAX    64
#define TRI     2080            // N*(N+1)/2
#define N_NODES (BGR * 48)      // 98304
#define N_EDGES (BGR * 128)     // 262144
#define BN_EPS  1e-5f

// ---------------------------------------------------------------------------
// GEMM: C[M,N] = A[M,K] @ W[K,N] + bias[N] (fp32 memory, f16 WMMA compute).
// Requires M%64==0, K%32==0; N tail guarded branch-free (mapA: N=2080).
// Block = 256 threads (8 waves); block tile 64Mx64N; wave = 16Mx32N (2 acc).
//
// WMMA f16 fragment order (ISA 7.12.2): lane = (row/col & 15) + khalf*16,
// 16 f16 per lane; element idx for k: half=(k>>3)&1, idx=(k&7)+((k&16)>>1).
// LDS tiles are stored directly in this order -> fragment = 2x ds_load_b128.
// ---------------------------------------------------------------------------
__global__ __launch_bounds__(256) void gemm_bias_f16w(
    const float* __restrict__ A, const float* __restrict__ W,
    const float* __restrict__ bias, float* __restrict__ C,
    int M, int N, int K)
{
    __shared__ __align__(16) _Float16 sAf[4 * 32 * 16];  // 4 m-tiles x 32 lanes x 16
    __shared__ __align__(16) _Float16 sBf[4 * 32 * 16];  // 4 n-tiles x 32 lanes x 16

    const int t    = threadIdx.x;
    const int lane = t & 31;
    const int wave = t >> 5;
    const int mBlock = blockIdx.y * 64;
    const int nBlock = blockIdx.x * 64;
    const bool nFull = (nBlock + 64 <= N);

    // ---- A staging coords: thread -> (row 0..63, k-octet) ----
    const int ar = t >> 2;                 // 0..63
    const int ac = (t & 3) * 8;            // 0,8,16,24
    const int aHalf = (ac >> 3) & 1;
    const int aIdx0 = (ac & 16) >> 1;      // 0 or 8
    _Float16* aDst =
        &sAf[(((ar >> 4) * 32) + (ar & 15) + aHalf * 16) * 16 + aIdx0];

    // ---- B staging coords: thread -> (col 0..63, k-octet) ----
    const int bc = t & 63;                 // col in tile
    const int bq = t >> 6;                 // k octet 0..3
    const int bHalf = bq & 1;
    const int bIdx0 = (bq >> 1) * 8;
    _Float16* bDst =
        &sBf[(((bc >> 4) * 32) + (bc & 15) + bHalf * 16) * 16 + bIdx0];
    const int bCol  = nBlock + bc;
    const int bColC = (bCol < N) ? bCol : (N - 1);  // clamped (branch-free tail)

    // ---- per-wave fragment pointers ----
    const v16h* fragA  = (const v16h*)&sAf[(((wave & 3) * 32) + lane) * 16];
    const v16h* fragB0 = (const v16h*)&sBf[((((wave >> 2) * 2 + 0) * 32) + lane) * 16];
    const v16h* fragB1 = (const v16h*)&sBf[((((wave >> 2) * 2 + 1) * 32) + lane) * 16];

    v8f acc0 = {}, acc1 = {};

    for (int k0 = 0; k0 < K; k0 += 32) {
        // stage A (64 rows x 32 k): 2x b128 load, packed cvt, 1x ds_store_b128
        {
            const float* src = A + (size_t)(mBlock + ar) * K + k0 + ac;
            const float4 f0 = *(const float4*)(src);
            const float4 f1 = *(const float4*)(src + 4);
            v8h pk;
            pk[0] = (_Float16)f0.x; pk[1] = (_Float16)f0.y;
            pk[2] = (_Float16)f0.z; pk[3] = (_Float16)f0.w;
            pk[4] = (_Float16)f1.x; pk[5] = (_Float16)f1.y;
            pk[6] = (_Float16)f1.z; pk[7] = (_Float16)f1.w;
            *(v8h*)aDst = pk;
            if (k0 + 32 < K) __builtin_prefetch(src + 32, 0, 3);
        }
        // stage B (32 k x 64 cols): 8 coalesced b32 loads (transpose for free)
        {
            const float* src = W + (size_t)(k0 + bq * 8) * N;
            v8h pk;
            if (nFull) {
                #pragma unroll
                for (int j = 0; j < 8; ++j)
                    pk[j] = (_Float16)src[(size_t)j * N + bCol];
            } else {
                #pragma unroll
                for (int j = 0; j < 8; ++j) {
                    const float v = src[(size_t)j * N + bColC];
                    pk[j] = (bCol < N) ? (_Float16)v : (_Float16)0.0f;
                }
            }
            *(v8h*)bDst = pk;
            if (k0 + 32 < K)
                __builtin_prefetch(src + (size_t)32 * N + bColC, 0, 3);
        }
        __syncthreads();

        const v16h a  = *fragA;    // 2x ds_load_b128
        const v16h b0 = *fragB0;
        const v16h b1 = *fragB1;
        acc0 = __builtin_amdgcn_wmma_f32_16x16x32_f16(
                   false, a, false, b0, (short)0, acc0, false, false);
        acc1 = __builtin_amdgcn_wmma_f32_16x16x32_f16(
                   false, a, false, b1, (short)0, acc1, false, false);
        __syncthreads();
    }

    // epilogue: C/D layout -> VGPR r holds M=r (lanes<16) / r+8 (lanes>=16)
    const int rowB = mBlock + (wave & 3) * 16 + ((lane >> 4) << 3);
    const int col0 = nBlock + (wave >> 2) * 32 + (lane & 15);
    const int col1 = col0 + 16;
    if (col0 < N) {
        const float bv = bias[col0];
        #pragma unroll
        for (int r = 0; r < 8; ++r)
            C[(size_t)(rowB + r) * N + col0] = acc0[r] + bv;
    }
    if (col1 < N) {
        const float bv = bias[col1];
        #pragma unroll
        for (int r = 0; r < 8; ++r)
            C[(size_t)(rowB + r) * N + col1] = acc1[r] + bv;
    }
}

// ---------------------------------------------------------------------------
// Fused encoder head + pool:  s = clip(h@sW+sb), f = tanh(h@tW+tb)/(1+exp(s)),
// feat[batch[n]] += f.   h:[Nn,64], W:[64,512].  One wave -> 16x16 tile, K=64.
// ---------------------------------------------------------------------------
__global__ __launch_bounds__(32) void pool_st_wmma(
    const float* __restrict__ h,
    const float* __restrict__ sW, const float* __restrict__ sb,
    const float* __restrict__ tW, const float* __restrict__ tb,
    const int* __restrict__ batch, float* __restrict__ feat)
{
    const int nodeBase = blockIdx.y * 16;
    const int colBase  = blockIdx.x * 16;
    const int lane  = threadIdx.x;
    const int lrow  = lane & 15;
    const int khalf = (lane < 16) ? 0 : 8;

    v8f cs = {}, ct = {};
    #pragma unroll
    for (int k0 = 0; k0 < 64; k0 += 32) {
        v16h a, bs, bt;
        #pragma unroll
        for (int idx = 0; idx < 16; ++idx) {
            const int kk = k0 + khalf + ((idx < 8) ? idx : idx + 8);
            a[idx]  = (_Float16)h[(size_t)(nodeBase + lrow) * 64 + kk];
            bs[idx] = (_Float16)sW[(size_t)kk * 512 + colBase + lrow];
            bt[idx] = (_Float16)tW[(size_t)kk * 512 + colBase + lrow];
        }
        cs = __builtin_amdgcn_wmma_f32_16x16x32_f16(false, a, false, bs, (short)0, cs, false, false);
        ct = __builtin_amdgcn_wmma_f32_16x16x32_f16(false, a, false, bt, (short)0, ct, false, false);
    }

    const int col = colBase + lrow;
    const float sbv = sb[col], tbv = tb[col];
    #pragma unroll
    for (int r = 0; r < 8; ++r) {
        const int node = nodeBase + r + ((lane >> 4) << 3);
        float s = cs[r] + sbv;
        s = fminf(fmaxf(s, -30.0f), 30.0f);
        const float f = tanhf(ct[r] + tbv) / (1.0f + __expf(s));
        atomicAdd(&feat[(size_t)batch[node] * 512 + col], f);
    }
}

// ---------------------------------------------------------------------------
// NNConv edge messages: msg[e,o] = sum_i h[src,i]*(sum_k ea[e,k]*nnW[k,i*D+o]
//                                                   + nnb[i*D+o]);
// agg[dst] += msg.  One wave per edge; lane -> output channel(s).
// ---------------------------------------------------------------------------
template<int DIN, int DOUT>
__global__ __launch_bounds__(256) void edge_msg(
    const float* __restrict__ h, const int* __restrict__ ei,
    const float* __restrict__ ea, const float* __restrict__ nnW,
    const float* __restrict__ nnb, float* __restrict__ agg, int E)
{
    const int wave = (int)((blockIdx.x * (size_t)blockDim.x + threadIdx.x) >> 5);
    const int lane = threadIdx.x & 31;
    if (wave >= E) return;

    const int src = ei[wave];
    const int dst = ei[E + wave];

    float e8[8];
    #pragma unroll
    for (int k = 0; k < 8; ++k) e8[k] = ea[(size_t)wave * 8 + k];
    float hs[DIN];
    #pragma unroll
    for (int i = 0; i < DIN; ++i) hs[i] = h[(size_t)src * DIN + i];

    constexpr int OO = DOUT / 32;
    float acc[OO];
    #pragma unroll
    for (int oo = 0; oo < OO; ++oo) acc[oo] = 0.0f;

    for (int i = 0; i < DIN; ++i) {
        const float hv = hs[i];
        #pragma unroll
        for (int oo = 0; oo < OO; ++oo) {
            const int o = lane + oo * 32;
            float w = nnb[i * DOUT + o];
            #pragma unroll
            for (int k = 0; k < 8; ++k)
                w += e8[k] * nnW[(size_t)k * (DIN * DOUT) + i * DOUT + o];
            acc[oo] += hv * w;
        }
    }
    #pragma unroll
    for (int oo = 0; oo < OO; ++oo)
        atomicAdd(&agg[(size_t)dst * DOUT + lane + oo * 32], acc[oo]);
}

// y = agg + h@root + bias
template<int DIN>
__global__ void node_update(const float* __restrict__ h, const float* __restrict__ agg,
                            const float* __restrict__ root, const float* __restrict__ bias,
                            float* __restrict__ y, int Nn, int DOUT)
{
    const size_t idx = blockIdx.x * (size_t)blockDim.x + threadIdx.x;
    if (idx >= (size_t)Nn * DOUT) return;
    const int n = (int)(idx / DOUT), o = (int)(idx % DOUT);
    float acc = agg[idx] + bias[o];
    #pragma unroll
    for (int i = 0; i < DIN; ++i)
        acc += h[(size_t)n * DIN + i] * root[i * DOUT + o];
    y[idx] = acc;
}

// Column sum / sumsq over rows (BatchNorm batch stats), LDS-atomic reduce.
__global__ __launch_bounds__(256) void bn_stats(const float* __restrict__ y,
                                                int rows, int C,
                                                float* __restrict__ gsum,
                                                float* __restrict__ gsq)
{
    __shared__ float ssum[2048];
    __shared__ float ssq[2048];
    for (int c = threadIdx.x; c < C; c += 256) { ssum[c] = 0.0f; ssq[c] = 0.0f; }
    __syncthreads();
    const size_t total = (size_t)rows * C;
    for (size_t i = blockIdx.x * (size_t)256 + threadIdx.x; i < total;
         i += (size_t)gridDim.x * 256) {
        const float v = y[i];
        const int c = (int)(i % C);
        atomicAdd(&ssum[c], v);
        atomicAdd(&ssq[c], v * v);
    }
    __syncthreads();
    for (int c = threadIdx.x; c < C; c += 256) {
        atomicAdd(&gsum[c], ssum[c]);
        atomicAdd(&gsq[c], ssq[c]);
    }
}

__global__ void bn_apply_relu(const float* __restrict__ y,
                              const float* __restrict__ gsum, const float* __restrict__ gsq,
                              const float* __restrict__ gamma, const float* __restrict__ beta,
                              float* __restrict__ out, int rows, int C)
{
    const size_t idx = blockIdx.x * (size_t)blockDim.x + threadIdx.x;
    if (idx >= (size_t)rows * C) return;
    const int c = (int)(idx % C);
    const float m = gsum[c] / rows;
    const float v = gsq[c] / rows - m * m;
    const float r = (y[idx] - m) * rsqrtf(v + BN_EPS) * gamma[c] + beta[c];
    out[idx] = fmaxf(r, 0.0f);
}

// z = eps * exp(0.5*lsgms) + mu
__global__ void reparam(const float* __restrict__ eps, const float* __restrict__ mu,
                        const float* __restrict__ ls, float* __restrict__ z, int n)
{
    const int i = blockIdx.x * blockDim.x + threadIdx.x;
    if (i < n) z[i] = eps[i] * __expf(0.5f * ls[i]) + mu[i];
}

// A_full[b,i,j] = A_full[b,j,i] = sigmoid(Apre[b, triu(i,j)])
__global__ void adj_recover(const float* __restrict__ Apre, float* __restrict__ Afull)
{
    const size_t idx = blockIdx.x * (size_t)blockDim.x + threadIdx.x;
    if (idx >= (size_t)BGR * TRI) return;
    const int b = (int)(idx / TRI);
    int rem = (int)(idx % TRI);
    int i = 0;
    while (rem >= (NMAX - i)) { rem -= (NMAX - i); ++i; }
    const int j = i + rem;
    const float s = 1.0f / (1.0f + __expf(-Apre[idx]));
    float* base = Afull + (size_t)b * (NMAX * NMAX);
    base[i * NMAX + j] = s;
    base[j * NMAX + i] = s;
}

__global__ void fill_zero(float* __restrict__ p, size_t n)
{
    const size_t i = blockIdx.x * (size_t)blockDim.x + threadIdx.x;
    if (i < n) p[i] = 0.0f;
}

// ---------------------------------------------------------------------------
static inline void zero_async(float* p, size_t n, hipStream_t s)
{
    fill_zero<<<dim3((unsigned)((n + 255) / 256)), dim3(256), 0, s>>>(p, n);
}
static inline void gemm(const float* A, const float* W, const float* b, float* C,
                        int M, int N, int K, hipStream_t s)
{
    gemm_bias_f16w<<<dim3((N + 63) / 64, M / 64), dim3(256), 0, s>>>(A, W, b, C, M, N, K);
}

extern "C" void kernel_launch(void* const* d_in, const int* in_sizes, int n_in,
                              void* d_out, int out_size, void* d_ws, size_t ws_size,
                              hipStream_t stream)
{
    (void)in_sizes; (void)n_in; (void)out_size; (void)ws_size;

    // ---- inputs (JAX pytree flatten: dict keys sorted) ----
    const float* x    = (const float*)d_in[0];
    const int*   ei   = (const int*)  d_in[1];
    const float* ea   = (const float*)d_in[2];
    const int*   bat  = (const int*)  d_in[3];
    /* d_in[4] edge_batch unused */
    const float* eps  = (const float*)d_in[5];
    // enc: convs[0]{beta,bias,gamma,nnW,nnb,root}, convs[1]{..}, s_lin, t_lin
    const float* c0_beta = (const float*)d_in[6];
    const float* c0_bias = (const float*)d_in[7];
    const float* c0_gam  = (const float*)d_in[8];
    const float* c0_nnW  = (const float*)d_in[9];
    const float* c0_nnb  = (const float*)d_in[10];
    const float* c0_root = (const float*)d_in[11];
    const float* c1_beta = (const float*)d_in[12];
    const float* c1_bias = (const float*)d_in[13];
    const float* c1_gam  = (const float*)d_in[14];
    const float* c1_nnW  = (const float*)d_in[15];
    const float* c1_nnb  = (const float*)d_in[16];
    const float* c1_root = (const float*)d_in[17];
    const float* sW = (const float*)d_in[18]; const float* sb = (const float*)d_in[19];
    const float* tW = (const float*)d_in[20]; const float* tb = (const float*)d_in[21];
    // dec: e11, e12, lins[i]{W,b,beta,gamma}, mapA, mapE, mapX
    const float* e11W = (const float*)d_in[22]; const float* e11b = (const float*)d_in[23];
    const float* e12W = (const float*)d_in[24]; const float* e12b = (const float*)d_in[25];
    const float* l0W = (const float*)d_in[26]; const float* l0b = (const float*)d_in[27];
    const float* l0be = (const float*)d_in[28]; const float* l0g = (const float*)d_in[29];
    const float* l1W = (const float*)d_in[30]; const float* l1b = (const float*)d_in[31];
    const float* l1be = (const float*)d_in[32]; const float* l1g = (const float*)d_in[33];
    const float* l2W = (const float*)d_in[34]; const float* l2b = (const float*)d_in[35];
    const float* l2be = (const float*)d_in[36]; const float* l2g = (const float*)d_in[37];
    const float* mAW = (const float*)d_in[38]; const float* mAb = (const float*)d_in[39];
    const float* mEW = (const float*)d_in[40]; const float* mEb = (const float*)d_in[41];
    const float* mXW = (const float*)d_in[42]; const float* mXb = (const float*)d_in[43];

    // ---- outputs (tuple concat: x_rec, A_full, E, z_mu, z_lsgms) ----
    float* out      = (float*)d_out;
    float* out_xrec = out;                               // 2048*64*32
    float* out_A    = out + (size_t)4194304;             // 2048*64*64
    float* out_E    = out + (size_t)12582912;            // 2048*2080*8
    float* out_zmu  = out + (size_t)46661632;            // 2048*256
    float* out_zls  = out + (size_t)47185920;            // 2048*256

    // ---- workspace layout (floats) ----
    float* ws    = (float*)d_ws;
    float* agg   = ws;                       // 98304*64
    float* ybuf  = ws + (size_t)6291456;     // 98304*64
    float* h1    = ws + (size_t)12582912;    // 98304*32
    float* h2    = ws + (size_t)15728640;    // 98304*64
    float* feat  = ws + (size_t)22020096;    // 2048*512
    float* z     = ws + (size_t)23068672;    // 2048*256
    float* zb0   = ws + (size_t)23592960;    // 2048*512
    float* zb1   = ws + (size_t)24641536;    // 2048*1024
    float* zb2   = ws + (size_t)26738688;    // 2048*2048
    float* Apre  = ws + (size_t)30932992;    // 2048*2080
    float* gsum  = ws + (size_t)35192832;    // 2048
    float* gsq   = gsum + 2048;              // 2048

    const int Nn = N_NODES, E = N_EDGES;
    const dim3 b256(256);

    // ================= encoder: NNConv layer 1 (32 -> 32) =================
    zero_async(agg, (size_t)Nn * 32, stream);
    edge_msg<32, 32><<<dim3(E / 8), b256, 0, stream>>>(x, ei, ea, c0_nnW, c0_nnb, agg, E);
    node_update<32><<<dim3((unsigned)(((size_t)Nn * 32 + 255) / 256)), b256, 0, stream>>>(
        x, agg, c0_root, c0_bias, ybuf, Nn, 32);
    zero_async(gsum, 4096, stream);
    bn_stats<<<dim3(1024), b256, 0, stream>>>(ybuf, Nn, 32, gsum, gsq);
    bn_apply_relu<<<dim3((unsigned)(((size_t)Nn * 32 + 255) / 256)), b256, 0, stream>>>(
        ybuf, gsum, gsq, c0_gam, c0_beta, h1, Nn, 32);

    // ================= encoder: NNConv layer 2 (32 -> 64) =================
    zero_async(agg, (size_t)Nn * 64, stream);
    edge_msg<32, 64><<<dim3(E / 8), b256, 0, stream>>>(h1, ei, ea, c1_nnW, c1_nnb, agg, E);
    node_update<32><<<dim3((unsigned)(((size_t)Nn * 64 + 255) / 256)), b256, 0, stream>>>(
        h1, agg, c1_root, c1_bias, ybuf, Nn, 64);
    zero_async(gsum, 4096, stream);
    bn_stats<<<dim3(1024), b256, 0, stream>>>(ybuf, Nn, 64, gsum, gsq);
    bn_apply_relu<<<dim3((unsigned)(((size_t)Nn * 64 + 255) / 256)), b256, 0, stream>>>(
        ybuf, gsum, gsq, c1_gam, c1_beta, h2, Nn, 64);

    // ============ fused s/t heads + global_add_pool (WMMA) ============
    zero_async(feat, (size_t)BGR * 512, stream);
    pool_st_wmma<<<dim3(512 / 16, Nn / 16), dim3(32), 0, stream>>>(
        h2, sW, sb, tW, tb, bat, feat);

    // ================= decoder =================
    gemm(feat, e11W, e11b, out_zmu, BGR, 256, 512, stream);
    gemm(feat, e12W, e12b, out_zls, BGR, 256, 512, stream);
    reparam<<<dim3((BGR * 256 + 255) / 256), b256, 0, stream>>>(
        eps, out_zmu, out_zls, z, BGR * 256);

    gemm(z, l0W, l0b, zb0, BGR, 512, 256, stream);
    zero_async(gsum, 4096, stream);
    bn_stats<<<dim3(1024), b256, 0, stream>>>(zb0, BGR, 512, gsum, gsq);
    bn_apply_relu<<<dim3((BGR * 512 + 255) / 256), b256, 0, stream>>>(
        zb0, gsum, gsq, l0g, l0be, zb0, BGR, 512);

    gemm(zb0, l1W, l1b, zb1, BGR, 1024, 512, stream);
    zero_async(gsum, 4096, stream);
    bn_stats<<<dim3(1024), b256, 0, stream>>>(zb1, BGR, 1024, gsum, gsq);
    bn_apply_relu<<<dim3((BGR * 1024 + 255) / 256), b256, 0, stream>>>(
        zb1, gsum, gsq, l1g, l1be, zb1, BGR, 1024);

    gemm(zb1, l2W, l2b, zb2, BGR, 2048, 1024, stream);
    zero_async(gsum, 4096, stream);
    bn_stats<<<dim3(1024), b256, 0, stream>>>(zb2, BGR, 2048, gsum, gsq);
    bn_apply_relu<<<dim3((unsigned)(((size_t)BGR * 2048 + 255) / 256)), b256, 0, stream>>>(
        zb2, gsum, gsq, l2g, l2be, zb2, BGR, 2048);

    // heads
    gemm(zb2, mXW, mXb, out_xrec, BGR, 2048, 2048, stream);   // x_rec
    gemm(zb2, mAW, mAb, Apre,     BGR, 2080, 2048, stream);   // A (pre-sigmoid)
    gemm(zb2, mEW, mEb, out_E,    BGR, 16640, 2048, stream);  // E

    // adjacency recovery: sigmoid triu -> symmetric full
    adj_recover<<<dim3((unsigned)(((size_t)BGR * TRI + 255) / 256)), b256, 0, stream>>>(
        Apre, out_A);
}
